// SiameseGNN_1692217115164
// MI455X (gfx1250) — matrix operand order repaired and checked by reading, not compiled, not run on hardware
//
#include <hip/hip_runtime.h>
#include <math.h>

// SiameseGNN for MI455X (gfx1250, wave32, WMMA bf16 16x16x32, f32 accum).
// B=4096, M=16, H=128, L=2. Output: d_out[0]=loss, d_out[1..]=sim[4096*4096].
// Deterministic: no float atomics; cross-block reductions via partial buffers.
// All WMMA operands are pre-packed bf16 (weights in fragment-native order,
// activations staged packed in LDS / stored packed in ws), so the inner loops
// are pure {ds_load_b128 / global_load_b128 -> v_wmma} with no conversion VALU.

#define Bn 4096
#define Mn 16
#define Hn 128
#define TAUv 14.2857142857142857f
#define BNEPS 1e-5f

typedef __attribute__((ext_vector_type(16))) __bf16 v16bf;
typedef __attribute__((ext_vector_type(8)))  float v8f;
typedef __attribute__((ext_vector_type(8)))  unsigned int v8u;

static __device__ __forceinline__ unsigned bf2pk(float lo, float hi) {
  unsigned a = __float_as_uint(lo);
  unsigned b = __float_as_uint(hi);
  a = (a + 0x7FFFu + ((a >> 16) & 1u)) >> 16;   // round-to-nearest-even bf16
  b = (b + 0x7FFFu + ((b >> 16) & 1u)) >> 16;
  return (a & 0xFFFFu) | ((b & 0xFFFFu) << 16);
}

static __device__ __forceinline__ v16bf frag_ld(const unsigned* p0,
                                                const unsigned* p1) {
  uint4 a = *(const uint4*)p0;
  uint4 b = *(const uint4*)p1;
  v8u t;
  t[0] = a.x; t[1] = a.y; t[2] = a.z; t[3] = a.w;
  t[4] = b.x; t[5] = b.y; t[6] = b.z; t[7] = b.w;
  return __builtin_bit_cast(v16bf, t);
}

// A fragment from packed-bf16 row-major tile (ldw dwords per 16-elem row).
// 16-bit A layout (contiguous k): lane<16 -> K 0..15; lane>=16 -> K 16..31.
static __device__ __forceinline__ v16bf frag_a_p(const unsigned* sAp, int ldw) {
  int lane = threadIdx.x & 31;
  const unsigned* r = sAp + (lane & 15) * ldw + ((lane < 16) ? 0 : 8);
  return frag_ld(r, r + 4);
}

// B fragment from weight packed in fragment-native order [kt][nt][lane][8].
static __device__ __forceinline__ v16bf frag_b_p(const unsigned* Wp, int kt,
                                                 int nt) {
  int lane = threadIdx.x & 31;
  const unsigned* p = Wp + (((kt * 8 + nt) * 32 + lane) << 3);
  return frag_ld(p, p + 4);
}

// B^T fragment from packed row-major activations: B[k][n] = row(n)[k].
// Vrow points at row (j0+n0), ldw dwords per row, already offset by k0/2.
static __device__ __forceinline__ v16bf frag_bt_p(const unsigned* Vrow, int ldw) {
  int lane = threadIdx.x & 31;
  const unsigned* p = Vrow + (lane & 15) * ldw + ((lane < 16) ? 0 : 8);
  return frag_ld(p, p + 4);
}

static __device__ __forceinline__ v8f wmma_bf16(v16bf a, v16bf b, v8f c) {
  return __builtin_amdgcn_wmma_f32_16x16x32_bf16(false, a, false, b, (short)0, c,
                                                 false, false);
}

// 16xK (packed LDS) @ KxHn (packed weight) -> f32 dst (row-major, stride ldd).
// 8 waves; wave w owns cols [16w,16w+16). act: 0 none, 1 relu, 2 tanh.
static __device__ __forceinline__ void block_gemm_p(const unsigned* sAp, int K,
    const unsigned* Wp, const float* bias, float* dst, int ldd, int act) {
  int nt = threadIdx.x >> 5;
  int ldw = K >> 1;
  v8f acc = {};
  for (int kt = 0; kt < (K >> 5); ++kt)
    acc = wmma_bf16(frag_a_p(sAp + kt * 16, ldw), frag_b_p(Wp, kt, nt), acc);
  int lane = threadIdx.x & 31;
  int n = nt * 16 + (lane & 15);
  int mb = (lane < 16) ? 0 : 8;
  float bb = bias ? bias[n] : 0.f;
#pragma unroll
  for (int r = 0; r < 8; ++r) {
    float v = acc[r] + bb;
    if (act == 1) v = fmaxf(v, 0.f);
    else if (act == 2) v = tanhf(v);
    dst[(mb + r) * ldd + n] = v;
  }
}

// Pack f32 weight [K,128] into bf16 fragment-native order (K*64 dwords).
__global__ __launch_bounds__(256) void packw_kernel(const float* __restrict__ W,
    unsigned* __restrict__ Wp, int K) {
  int total = K * 64;
  for (int idx = blockIdx.x * 256 + threadIdx.x; idx < total;
       idx += gridDim.x * 256) {
    int i = idx & 7;
    int lane = (idx >> 3) & 31;
    int nt = (idx >> 8) & 7;
    int kt = idx >> 11;
    int n = nt * 16 + (lane & 15);
    int k = kt * 32 + ((lane < 16) ? 0 : 16) + 2 * i;
    Wp[idx] = bf2pk(W[k * Hn + n], W[(k + 1) * Hn + n]);
  }
}

// ---------------- GNN layer kernel -------------------------------------
// One block = 16 patients. layer0: m_fwd (->aggm partials) + m_bwd + xp update.
// layer1: edge-attr recompute on the fly + m_bwd + xp update (m_fwd is dead).
__global__ __launch_bounds__(256) void g1_kernel(
    const float* __restrict__ x, const int* __restrict__ mask,
    const float* __restrict__ xp_in, const float* __restrict__ xm_in,
    const unsigned* __restrict__ Wmp, const float* __restrict__ bm,
    const float* __restrict__ Wa, const float* __restrict__ ba,
    const unsigned* __restrict__ Wep, const float* __restrict__ be,
    float* __restrict__ xp_out, float* __restrict__ aggm_part, int layer) {
  __shared__ __align__(16) unsigned sAp[Mn * 192];  // packed A, max K=384
  __shared__ float sMsg[Mn * Hn];
  __shared__ float sAttr[Mn * Hn];
  __shared__ float sAgg[Mn * Hn];
  __shared__ float sVec[2 * Hn];
  __shared__ float sRed[256];
  __shared__ float sMask[Mn];
  const int tid = threadIdx.x;

  for (int i = tid; i < Mn * Hn; i += 256) sAgg[i] = 0.f;

  for (int bi = 0; bi < 16; ++bi) {
    int b = blockIdx.x * 16 + bi;
    if (tid < Mn) sMask[tid] = (mask[b * Mn + tid] != 0) ? 1.f : 0.f;
    __syncthreads();
    float cntp = 0.f;
#pragma unroll
    for (int m = 0; m < Mn; ++m) cntp += sMask[m];
    cntp = fmaxf(cntp, 1.f);

    if (layer == 0) {
      // m_fwd: A row m = [ones(H) | x(b,m)*ones(H)]  (K=256, 128 dwords/row)
      for (int i = tid; i < Mn * 128; i += 256) {
        int r = i >> 7, cd = i & 127;
        float xv = x[b * Mn + r];
        sAp[i] = (cd < 64) ? bf2pk(1.f, 1.f) : bf2pk(xv, xv);
      }
      __syncthreads();
      block_gemm_p(sAp, 256, Wmp, bm, sMsg, Hn, 1);
      __syncthreads();
      for (int i = tid; i < Mn * Hn; i += 256) sAgg[i] += sMask[i >> 7] * sMsg[i];
      // m_bwd: A row m = [xm[m] | x(b,m)*ones(H)]
      for (int i = tid; i < Mn * 128; i += 256) {
        int r = i >> 7, cd = i & 127;
        unsigned v;
        if (cd < 64) {
          int c = 2 * cd;
          v = bf2pk(xm_in[r * Hn + c], xm_in[r * Hn + c + 1]);
        } else {
          float xv = x[b * Mn + r];
          v = bf2pk(xv, xv);
        }
        sAp[i] = v;
      }
      __syncthreads();
      block_gemm_p(sAp, 256, Wmp, bm, sMsg, Hn, 1);
      __syncthreads();
    } else {
      // attr_b = relu([xm[m] | xp[b] | x(b,m)] @ We + be)  (K=384)
      for (int i = tid; i < Mn * 192; i += 256) {
        int r = i / 192, cd = i % 192;
        unsigned v;
        if (cd < 64) {
          int c = 2 * cd;
          v = bf2pk(xm_in[r * Hn + c], xm_in[r * Hn + c + 1]);
        } else if (cd < 128) {
          int c = 2 * cd - 256 + Hn;  // = 2*(cd-64)
          v = bf2pk(xp_in[b * Hn + c], xp_in[b * Hn + c + 1]);
        } else {
          float xv = x[b * Mn + r];
          v = bf2pk(xv, xv);
        }
        sAp[i] = v;
      }
      __syncthreads();
      block_gemm_p(sAp, 384, Wep, be, sAttr, Hn, 1);
      __syncthreads();
      // m_bwd: A row m = [xm[m] | attr_b(m)]  (K=256)
      for (int i = tid; i < Mn * 128; i += 256) {
        int r = i >> 7, cd = i & 127;
        unsigned v;
        if (cd < 64) {
          int c = 2 * cd;
          v = bf2pk(xm_in[r * Hn + c], xm_in[r * Hn + c + 1]);
        } else {
          int c = 2 * cd - 128;
          v = bf2pk(sAttr[r * Hn + c], sAttr[r * Hn + c + 1]);
        }
        sAp[i] = v;
      }
      __syncthreads();
      block_gemm_p(sAp, 256, Wmp, bm, sMsg, Hn, 1);
      __syncthreads();
    }

    // aggr_p = masked-mean over m, then xp = relu([aggr_p|xp] @ Wa + ba)
    int h = tid & 127, half = tid >> 7;
    float s = 0.f;
#pragma unroll
    for (int j = 0; j < 8; ++j) {
      int m = half * 8 + j;
      s += sMask[m] * sMsg[m * Hn + h];
    }
    sRed[tid] = s;
    __syncthreads();
    if (half == 0) {
      sVec[h] = (sRed[tid] + sRed[tid + 128]) / cntp;
      sVec[Hn + h] = xp_in ? xp_in[b * Hn + h] : 1.f;
    }
    __syncthreads();
    float s2 = 0.f;
    for (int k = half * 128; k < half * 128 + 128; ++k)
      s2 += sVec[k] * Wa[k * Hn + h];
    sRed[tid] = s2;
    __syncthreads();
    if (half == 0)
      xp_out[b * Hn + h] = fmaxf(sRed[tid] + sRed[tid + 128] + ba[h], 0.f);
    __syncthreads();
  }

  if (layer == 0)
    for (int i = tid; i < Mn * Hn; i += 256)
      aggm_part[blockIdx.x * (Mn * Hn) + i] = sAgg[i];
}

// Reduce aggm partials, compute cnt_m, xm = relu([aggm/cnt|xm] @ Wa + ba).
__global__ __launch_bounds__(256) void g2_kernel(
    const int* __restrict__ mask, const float* __restrict__ xm_in,
    const unsigned* __restrict__ Wap, const float* __restrict__ ba,
    const float* __restrict__ part, int nparts, float* __restrict__ xm_out) {
  __shared__ __align__(16) unsigned sAp[Mn * 128];
  __shared__ float sRed[256];
  __shared__ float sCnt[Mn];
  int tid = threadIdx.x;
  int m = tid & 15, g = tid >> 4;
  float c = 0.f;
  for (int b = g; b < Bn; b += 16) c += (mask[b * Mn + m] != 0) ? 1.f : 0.f;
  sRed[tid] = c;
  __syncthreads();
  for (int off = 8; off >= 1; off >>= 1) {
    if (g < off) sRed[tid] += sRed[tid + off * 16];
    __syncthreads();
  }
  if (tid < Mn) sCnt[tid] = fmaxf(sRed[tid], 1.f);
  __syncthreads();
  for (int i = tid; i < Mn * 128; i += 256) {
    int r = i >> 7, cd = i & 127;
    unsigned v;
    if (cd < 64) {
      int c0 = r * Hn + 2 * cd;
      float s0 = 0.f, s1 = 0.f;
      for (int gg = 0; gg < nparts; ++gg) {
        const float* pp = part + gg * (Mn * Hn) + c0;
        s0 += pp[0];
        s1 += pp[1];
      }
      float inv = 1.f / sCnt[r];
      v = bf2pk(s0 * inv, s1 * inv);
    } else {
      int c0 = r * Hn + 2 * cd - 128;
      v = bf2pk(xm_in[c0], xm_in[c0 + 1]);
    }
    sAp[i] = v;
  }
  __syncthreads();
  block_gemm_p(sAp, 256, Wap, ba, xm_out, Hn, 1);
}

// out = act(X @ W + b); act 2 = tanh + row L2-normalize (cl_proj), also
// emits a packed-bf16 copy (outp) for downstream WMMA consumers.
__global__ __launch_bounds__(256) void mlp_kernel(
    const float* __restrict__ X, const unsigned* __restrict__ Wp,
    const float* __restrict__ bias, float* __restrict__ out,
    unsigned* __restrict__ outp, int act) {
  __shared__ __align__(16) unsigned sAp[16 * 64];
  __shared__ float sO[16 * Hn];
  __shared__ float sRed[256];
  int tid = threadIdx.x;
  int r0 = blockIdx.x * 16;
  for (int i = tid; i < 16 * 64; i += 256)
    sAp[i] = bf2pk(X[r0 * Hn + 2 * i], X[r0 * Hn + 2 * i + 1]);
  __syncthreads();
  block_gemm_p(sAp, Hn, Wp, bias, sO, Hn, act);
  __syncthreads();
  if (act == 2) {
    int r = tid >> 4, i = tid & 15;
    float s = 0.f;
    for (int c = i * 8; c < i * 8 + 8; ++c) { float v = sO[r * Hn + c]; s += v * v; }
    sRed[tid] = s;
    __syncthreads();
    for (int off = 8; off >= 1; off >>= 1) {
      if (i < off) sRed[tid] += sRed[tid + off];
      __syncthreads();
    }
    float nrm = fmaxf(sqrtf(sRed[r * 16]), 1e-12f);
    for (int c = i * 8; c < i * 8 + 8; c += 2) {
      float v0 = sO[r * Hn + c] / nrm;
      float v1 = sO[r * Hn + c + 1] / nrm;
      out[(r0 + r) * Hn + c] = v0;
      out[(r0 + r) * Hn + c + 1] = v1;
      outp[(r0 + r) * 64 + (c >> 1)] = bf2pk(v0, v1);
    }
  } else {
    for (int i = tid; i < 16 * Hn; i += 256) out[r0 * Hn + i] = sO[i];
  }
}

// lse[i] = logsumexp_j( TAU * dot(U_i, V_j) ), online over column tiles.
// Up/Vp are packed-bf16 [Bn][64] copies of the L2-normalized projections.
__global__ __launch_bounds__(256) void lse_kernel(
    const unsigned* __restrict__ Up, const unsigned* __restrict__ Vp,
    float* __restrict__ lse) {
  __shared__ __align__(16) unsigned sAp[16 * 64];
  __shared__ float sT[16 * Hn];
  __shared__ float sM[16], sS[16];
  __shared__ float sRed[256];
  int tid = threadIdx.x;
  int r0 = blockIdx.x * 16;
  for (int i = tid; i < 16 * 64; i += 256) sAp[i] = Up[r0 * 64 + i];
  if (tid < 16) { sM[tid] = -3.0e38f; sS[tid] = 0.f; }
  __syncthreads();
  int lane = tid & 31, n0 = (tid >> 5) * 16;
  for (int j0 = 0; j0 < Bn; j0 += 128) {
    if (j0 + 128 < Bn)  // pull next tile's rows toward L2/L0 (global_prefetch)
      __builtin_prefetch(Vp + (j0 + 128 + n0 + (lane & 15)) * 64, 0, 0);
    v8f acc = {};
    const unsigned* vrow = Vp + (j0 + n0) * 64;
#pragma unroll
    for (int kt = 0; kt < 4; ++kt)
      acc = wmma_bf16(frag_a_p(sAp + kt * 16, 64), frag_bt_p(vrow + kt * 16, 64),
                      acc);
    int n = n0 + (lane & 15), mb = (lane < 16) ? 0 : 8;
#pragma unroll
    for (int r = 0; r < 8; ++r) sT[(mb + r) * Hn + n] = acc[r] * TAUv;
    __syncthreads();
    int rr = tid >> 4, ii = tid & 15;
    float lm = -3.0e38f;
    for (int c = ii * 8; c < ii * 8 + 8; ++c) lm = fmaxf(lm, sT[rr * Hn + c]);
    sRed[tid] = lm;
    __syncthreads();
    for (int off = 8; off >= 1; off >>= 1) {
      if (ii < off) sRed[tid] = fmaxf(sRed[tid], sRed[tid + off]);
      __syncthreads();
    }
    float tmax = sRed[rr * 16];
    __syncthreads();
    float ls = 0.f;
    for (int c = ii * 8; c < ii * 8 + 8; ++c) ls += __expf(sT[rr * Hn + c] - tmax);
    sRed[tid] = ls;
    __syncthreads();
    for (int off = 8; off >= 1; off >>= 1) {
      if (ii < off) sRed[tid] += sRed[tid + off];
      __syncthreads();
    }
    if (ii == 0) {
      float ts = sRed[rr * 16];
      float Mo = sM[rr];
      float nm = fmaxf(Mo, tmax);
      sS[rr] = sS[rr] * __expf(Mo - nm) + ts * __expf(tmax - nm);
      sM[rr] = nm;
    }
    __syncthreads();
  }
  if (tid < 16) lse[r0 + tid] = sM[tid] + __logf(sS[tid]);
}

__global__ __launch_bounds__(256) void loss_kernel(
    const float* __restrict__ u, const float* __restrict__ au,
    const float* __restrict__ lseR, const float* __restrict__ lseC,
    float* __restrict__ out) {
  __shared__ float sRed[256];
  int tid = threadIdx.x;
  float acc = 0.f;
  for (int i = tid; i < Bn; i += 256) {
    float d = 0.f;
    for (int k = 0; k < Hn; ++k) d += u[i * Hn + k] * au[i * Hn + k];
    acc += 0.5f * (lseR[i] + lseC[i]) - TAUv * d;
  }
  sRed[tid] = acc;
  __syncthreads();
  for (int off = 128; off >= 1; off >>= 1) {
    if (tid < off) sRed[tid] += sRed[tid + off];
    __syncthreads();
  }
  if (tid == 0) out[0] = sRed[0] / (float)Bn;
}

// Column sum / sum-of-squares partials (block g covers 128 rows).
__global__ __launch_bounds__(256) void stats_kernel(const float* __restrict__ X,
                                                    float* __restrict__ part) {
  int g = blockIdx.x, tid = threadIdx.x;
  int c = tid & 127, half = tid >> 7;
  float s = 0.f, q = 0.f;
  for (int r = g * 128 + half; r < g * 128 + 128; r += 2) {
    float v = X[r * Hn + c];
    s += v; q += v * v;
  }
  __shared__ float ls[256], lq[256];
  ls[tid] = s; lq[tid] = q;
  __syncthreads();
  if (half == 0) {
    part[g * 256 + c] = ls[tid] + ls[tid + 128];
    part[g * 256 + 128 + c] = lq[tid] + lq[tid + 128];
  }
}

__global__ void statfin_kernel(const float* __restrict__ part,
                               float* __restrict__ mu, float* __restrict__ var) {
  int c = threadIdx.x;  // 128 threads
  float s = 0.f, q = 0.f;
  for (int g = 0; g < 32; ++g) { s += part[g * 256 + c]; q += part[g * 256 + 128 + c]; }
  float m = s / (float)Bn;
  mu[c] = m;
  var[c] = q / (float)Bn - m * m;
}

// Batchnorm apply; writes f32 bn (for bandwidth colsums), packed bf16 copy
// (for the Gram WMMAs) and per-row squared norms.
__global__ __launch_bounds__(256) void bnapply_kernel(
    const float* __restrict__ X, const float* __restrict__ mu,
    const float* __restrict__ var, const float* __restrict__ gamma,
    const float* __restrict__ beta, float* __restrict__ bn,
    unsigned* __restrict__ bnp, float* __restrict__ sq) {
  int b = blockIdx.x * 256 + threadIdx.x;
  float q = 0.f;
  for (int c = 0; c < Hn; c += 2) {
    float v0 = gamma[c] * (X[b * Hn + c] - mu[c]) * rsqrtf(var[c] + BNEPS) + beta[c];
    float v1 = gamma[c + 1] * (X[b * Hn + c + 1] - mu[c + 1]) *
                   rsqrtf(var[c + 1] + BNEPS) + beta[c + 1];
    bn[b * Hn + c] = v0;
    bn[b * Hn + c + 1] = v1;
    bnp[b * 64 + (c >> 1)] = bf2pk(v0, v1);
    q += v0 * v0 + v1 * v1;
  }
  sq[b] = q;
}

// bw: sum(L2) = 2*n*sum(sq) - 2*||colsum||^2 (no O(B^2) pass needed).
__global__ __launch_bounds__(256) void bw_kernel(const float* __restrict__ bn,
                                                 const float* __restrict__ sq,
                                                 float* __restrict__ bwout) {
  __shared__ float sRed[256];
  int tid = threadIdx.x;
  int c = tid & 127, half = tid >> 7;
  float cs = 0.f;
  for (int r = half; r < Bn; r += 2) cs += bn[r * Hn + c];
  sRed[tid] = cs;
  __syncthreads();
  float csq = 0.f;
  if (half == 0) { float t = sRed[tid] + sRed[tid + 128]; csq = t * t; }
  __syncthreads();
  sRed[tid] = csq;
  __syncthreads();
  for (int off = 128; off >= 1; off >>= 1) {
    if (tid < off) sRed[tid] += sRed[tid + off];
    __syncthreads();
  }
  float sum_cs2 = sRed[0];
  __syncthreads();
  float ss = 0.f;
  for (int i = tid; i < Bn; i += 256) ss += sq[i];
  sRed[tid] = ss;
  __syncthreads();
  for (int off = 128; off >= 1; off >>= 1) {
    if (tid < off) sRed[tid] += sRed[tid + off];
    __syncthreads();
  }
  if (tid == 0) {
    float sumsq = sRed[0];
    float sumL2 = 2.f * (float)Bn * sumsq - 2.f * sum_cs2;
    float bwv = sumL2 / (float)((long long)Bn * Bn - Bn);
    if (bwv < 0.001f) bwv = 1.f;
    bwout[0] = bwv * 0.5f;  // / kernel_mul^(kernel_num//2)
  }
}

// Fused similarity tile: both Grams (bn_s, bn_z) + Gaussian mixes -> sim.
__global__ __launch_bounds__(256) void sim_kernel(
    const unsigned* __restrict__ bnSp, const float* __restrict__ sqS,
    const unsigned* __restrict__ bnZp, const float* __restrict__ sqZ,
    const float* __restrict__ bw, const float* __restrict__ eps0,
    float* __restrict__ out) {
  __shared__ __align__(16) unsigned aS[16 * 64];
  __shared__ __align__(16) unsigned aZ[16 * 64];
  __shared__ float rS[16], rZ[16], cS[128], cZ[128];
  int tid = threadIdx.x;
  int i0 = (blockIdx.x >> 5) * 16;
  int j0 = (blockIdx.x & 31) * 128;
  for (int i = tid; i < 16 * 64; i += 256) {
    aS[i] = bnSp[i0 * 64 + i];
    aZ[i] = bnZp[i0 * 64 + i];
  }
  if (tid < 16) { rS[tid] = sqS[i0 + tid]; rZ[tid] = sqZ[i0 + tid]; }
  if (tid < 128) { cS[tid] = sqS[j0 + tid]; cZ[tid] = sqZ[j0 + tid]; }
  __syncthreads();
  int lane = tid & 31, n0 = (tid >> 5) * 16;
  const unsigned* srow = bnSp + (j0 + n0) * 64;
  const unsigned* zrow = bnZp + (j0 + n0) * 64;
  v8f accS = {};
  v8f accZ = {};
#pragma unroll
  for (int kt = 0; kt < 4; ++kt) {
    accS = wmma_bf16(frag_a_p(aS + kt * 16, 64), frag_bt_p(srow + kt * 16, 64), accS);
    accZ = wmma_bf16(frag_a_p(aZ + kt * 16, 64), frag_bt_p(zrow + kt * 16, 64), accZ);
  }
  float bw1 = bw[0], bw2 = bw[1];
  float sg = 1.f / (1.f + __expf(-eps0[0]));
  int n = n0 + (lane & 15), mb = (lane < 16) ? 0 : 8;
#pragma unroll
  for (int r = 0; r < 8; ++r) {
    int m = mb + r;
    float l2s = rS[m] + cS[n] - 2.f * accS[r];
    float l2z = rZ[m] + cZ[n] - 2.f * accZ[r];
    float k1 = (__expf(-l2s / bw1) + __expf(-l2s / (2.f * bw1)) +
                __expf(-l2s / (4.f * bw1))) * (1.f / 3.f);
    float k2 = (__expf(-l2z / bw2) + __expf(-l2z / (2.f * bw2)) +
                __expf(-l2z / (4.f * bw2))) * (1.f / 3.f);
    float v = ((1.f - sg) * k1 + sg) * k2;
    out[(size_t)(i0 + m) * Bn + (j0 + n)] = fminf(fmaxf(v, 0.f), 1.f);
  }
}

extern "C" void kernel_launch(void* const* d_in, const int* in_sizes, int n_in,
                              void* d_out, int out_size, void* d_ws, size_t ws_size,
                              hipStream_t stream) {
  (void)in_sizes; (void)n_in; (void)out_size; (void)ws_size;
  const float* x     = (const float*)d_in[0];
  const float* mod   = (const float*)d_in[1];
  const float* msgW0 = (const float*)d_in[2];
  const float* msgb0 = (const float*)d_in[3];
  const float* aggW0 = (const float*)d_in[4];
  const float* aggb0 = (const float*)d_in[5];
  const float* msgW1 = (const float*)d_in[6];
  const float* msgb1 = (const float*)d_in[7];
  const float* aggW1 = (const float*)d_in[8];
  const float* aggb1 = (const float*)d_in[9];
  const float* edgeW = (const float*)d_in[10];
  const float* edgeb = (const float*)d_in[11];
  const float* clW   = (const float*)d_in[12];
  const float* clb   = (const float*)d_in[13];
  const float* sW1   = (const float*)d_in[14];
  const float* sb1   = (const float*)d_in[15];
  const float* sW2   = (const float*)d_in[16];
  const float* sb2   = (const float*)d_in[17];
  const float* sW3   = (const float*)d_in[18];
  const float* sb3   = (const float*)d_in[19];
  const float* gamma = (const float*)d_in[20];
  const float* beta  = (const float*)d_in[21];
  const float* eps0  = (const float*)d_in[22];
  const int*   xflag = (const int*)d_in[23];
  const int*   agflg = (const int*)d_in[24];

  float* ws = (float*)d_ws;
  size_t o = 0;
  auto alloc = [&](size_t nel) { float* p = ws + o; o += nel; return p; };
  float* xpA  = alloc((size_t)Bn * Hn);
  float* xmA  = alloc((size_t)Mn * Hn);
  float* part = alloc((size_t)256 * Mn * Hn);
  float* z    = alloc((size_t)Bn * Hn);
  float* az   = alloc((size_t)Bn * Hn);
  float* u    = alloc((size_t)Bn * Hn);
  float* au   = alloc((size_t)Bn * Hn);
  float* s1   = alloc((size_t)Bn * Hn);
  float* s2   = alloc((size_t)Bn * Hn);
  float* bnS  = alloc((size_t)Bn * Hn);
  float* bnZ  = alloc((size_t)Bn * Hn);
  float* sqS  = alloc(Bn);
  float* sqZ  = alloc(Bn);
  float* lseR = alloc(Bn);
  float* lseC = alloc(Bn);
  float* stat = alloc(32 * 256);
  float* mus  = alloc(Hn);
  float* vars = alloc(Hn);
  float* muz  = alloc(Hn);
  float* varz = alloc(Hn);
  float* bwv  = alloc(64);  // keep 256B alignment for following packed bufs
  unsigned* msgW0p = (unsigned*)alloc(256 * 64);
  unsigned* msgW1p = (unsigned*)alloc(256 * 64);
  unsigned* aggW0p = (unsigned*)alloc(256 * 64);
  unsigned* edgeWp = (unsigned*)alloc(384 * 64);
  unsigned* clWp   = (unsigned*)alloc(128 * 64);
  unsigned* sW1p   = (unsigned*)alloc(128 * 64);
  unsigned* sW2p   = (unsigned*)alloc(128 * 64);
  unsigned* sW3p   = (unsigned*)alloc(128 * 64);
  unsigned* up     = (unsigned*)alloc((size_t)Bn * 64);
  unsigned* aup    = (unsigned*)alloc((size_t)Bn * 64);
  unsigned* bnSp   = (unsigned*)alloc((size_t)Bn * 64);
  unsigned* bnZp   = (unsigned*)alloc((size_t)Bn * 64);

  float* outF = (float*)d_out;

  // Pack all WMMA weights once (fragment-native bf16).
  packw_kernel<<<64, 256, 0, stream>>>(msgW0, msgW0p, 256);
  packw_kernel<<<64, 256, 0, stream>>>(msgW1, msgW1p, 256);
  packw_kernel<<<64, 256, 0, stream>>>(aggW0, aggW0p, 256);
  packw_kernel<<<96, 256, 0, stream>>>(edgeW, edgeWp, 384);
  packw_kernel<<<32, 256, 0, stream>>>(clW, clWp, 128);
  packw_kernel<<<32, 256, 0, stream>>>(sW1, sW1p, 128);
  packw_kernel<<<32, 256, 0, stream>>>(sW2, sW2p, 128);
  packw_kernel<<<32, 256, 0, stream>>>(sW3, sW3p, 128);

  // GNN run 1 (x_flag) -> z
  g1_kernel<<<256, 256, 0, stream>>>(x, xflag, nullptr, mod, msgW0p, msgb0,
                                     aggW0, aggb0, nullptr, nullptr, xpA, part, 0);
  g2_kernel<<<1, 256, 0, stream>>>(xflag, mod, aggW0p, aggb0, part, 256, xmA);
  g1_kernel<<<256, 256, 0, stream>>>(x, xflag, xpA, xmA, msgW1p, msgb1,
                                     aggW1, aggb1, edgeWp, edgeb, z, part, 1);
  // GNN run 2 (ag_flag) -> az
  g1_kernel<<<256, 256, 0, stream>>>(x, agflg, nullptr, mod, msgW0p, msgb0,
                                     aggW0, aggb0, nullptr, nullptr, xpA, part, 0);
  g2_kernel<<<1, 256, 0, stream>>>(agflg, mod, aggW0p, aggb0, part, 256, xmA);
  g1_kernel<<<256, 256, 0, stream>>>(x, agflg, xpA, xmA, msgW1p, msgb1,
                                     aggW1, aggb1, edgeWp, edgeb, az, part, 1);

  // Contrastive head: cl_proj + online row/col LSE + loss (no BxB storage).
  mlp_kernel<<<256, 256, 0, stream>>>(z, clWp, clb, u, up, 2);
  mlp_kernel<<<256, 256, 0, stream>>>(az, clWp, clb, au, aup, 2);
  lse_kernel<<<256, 256, 0, stream>>>(up, aup, lseR);
  lse_kernel<<<256, 256, 0, stream>>>(aup, up, lseC);
  loss_kernel<<<1, 256, 0, stream>>>(u, au, lseR, lseC, outF);

  // Similarity head: 3-layer MLP, batchnorm, analytic bandwidth, fused sim.
  mlp_kernel<<<256, 256, 0, stream>>>(z, sW1p, sb1, s1, nullptr, 1);
  mlp_kernel<<<256, 256, 0, stream>>>(s1, sW2p, sb2, s2, nullptr, 1);
  mlp_kernel<<<256, 256, 0, stream>>>(s2, sW3p, sb3, s1, nullptr, 0);
  stats_kernel<<<32, 256, 0, stream>>>(s1, stat);
  statfin_kernel<<<1, 128, 0, stream>>>(stat, mus, vars);
  bnapply_kernel<<<16, 256, 0, stream>>>(s1, mus, vars, gamma, beta, bnS, bnSp, sqS);
  stats_kernel<<<32, 256, 0, stream>>>(z, stat);
  statfin_kernel<<<1, 128, 0, stream>>>(stat, muz, varz);
  bnapply_kernel<<<16, 256, 0, stream>>>(z, muz, varz, gamma, beta, bnZ, bnZp, sqZ);
  bw_kernel<<<1, 256, 0, stream>>>(bnS, sqS, bwv + 0);
  bw_kernel<<<1, 256, 0, stream>>>(bnZ, sqZ, bwv + 1);
  sim_kernel<<<8192, 256, 0, stream>>>(bnSp, sqS, bnZp, sqZ, bwv, eps0, outF + 1);
}
